// NystromAttention_31808527794676
// MI455X (gfx1250) — matrix operand order repaired and checked by reading, not compiled, not run on hardware
//
#include <hip/hip_runtime.h>

// ---------------------------------------------------------------------------
// Nystrom attention on MI455X (gfx1250), wave32 + v_wmma_f32_16x16x32_f16.
// All heavy GEMMs use WMMA with f16 operands / f32 accumulation; softmax and
// reductions in f32.  LDS tiles are double-buffered and filled with
// GLOBAL_LOAD_ASYNC_TO_LDS_B128 (ASYNCcnt) so the copy of tile s+1 overlaps
// the WMMA work on tile s.  Workspace requirement: ~166 MB.
// ---------------------------------------------------------------------------

#define DIMC   512
#define HEADS  8
#define HDIM   64
#define NSEQ   8192
#define NBATCH 2
#define BHN    16          // NBATCH*HEADS
#define NL     256         // landmarks
#define PITERS 6
#define SCALEF 0.125f      // 64^-0.5

typedef _Float16 f16;
typedef _Float16 v16h  __attribute__((ext_vector_type(16)));
typedef float    v8f   __attribute__((ext_vector_type(8)));
typedef uint32_t u32x4 __attribute__((ext_vector_type(4)));

union Frag {                 // one WMMA A or B operand (8 VGPRs)
    v16h     h;
    uint32_t u[8];
    u32x4    q[2];
};

__device__ __forceinline__ v8f wmma_f16(const Frag& a, const Frag& b, v8f c) {
    return __builtin_amdgcn_wmma_f32_16x16x32_f16(false, a.h, false, b.h,
                                                  (short)0, c, false, false);
}

// generic shared ptr -> 32-bit LDS byte address (flat addr[31:0] == LDS addr)
__device__ __forceinline__ uint32_t lds_addr(const void* p) {
    return (uint32_t)(uintptr_t)p;
}
// async 16B global -> LDS copy (per lane), tracked by ASYNCcnt
__device__ __forceinline__ void async_copy16(uint32_t lds, const f16* g) {
    asm volatile("global_load_async_to_lds_b128 %0, %1, off"
                 :: "v"(lds), "v"(g) : "memory");
}
__device__ __forceinline__ void wait_async0() {
    asm volatile("s_wait_asynccnt 0" ::: "memory");
}

// ---------------------------------------------------------------------------
// Generic tiled WMMA GEMM:  C[z] = A[z] (MxK, row-major, f16)
//                                 @ B[z]^T   with B given as:
//   BKN==false : B stored [N][K] row-major  -> async-staged to LDS
//   BKN==true  : B stored [K][N] row-major  -> transposed while staging (sync)
// Wave tile = 64x32 (4x2 WMMA tiles).  Block = (BM/64)*(BN/32) waves.
// Double-buffered K pipeline: one s_wait_asynccnt + one barrier per 32-step.
// EPI: 0 = store f16, 1 = store f16 of (2I - acc), 2 = f32 store + bias,
//      3 = store f16 transposed (C[col*ldc+row]), 5 = QKV scatter epilogue
//          (q scaled, v stored transposed [bh][hd][n]).
// ---------------------------------------------------------------------------
template<int BM, int BN, bool BKN, int EPI>
__global__ void __launch_bounds__((BM/64)*(BN/32)*32)
gemm_wmma(const f16* __restrict__ A, const f16* __restrict__ B,
          void* __restrict__ Cv, int K, int lda, int ldb, int ldc,
          long long sA, long long sB, long long sC,
          const float* __restrict__ bias,
          f16* __restrict__ q_out, f16* __restrict__ k_out, f16* __restrict__ v_out)
{
    constexpr int NW  = (BM/64)*(BN/32);
    constexpr int LDK = 40;                   // padded stride (80B, 16B aligned)
    static_assert(NW >= 1 && NW <= 8, "wave count");
    __shared__ alignas(16) f16 As[2][BM*LDK];
    __shared__ alignas(16) f16 Bs[2][BN*LDK];

    const int z = blockIdx.z;
    A += z * sA;  B += z * sB;
    const int tid  = threadIdx.x;
    const int wave = tid >> 5, lane = tid & 31, hf = lane >> 4, l16 = lane & 15;
    const int wm = wave % (BM/64), wn = wave / (BM/64);
    const int m0 = blockIdx.x * BM, n0 = blockIdx.y * BN;

    auto stageA = [&](int buf, int k0) {
        for (int u = tid; u < BM*4; u += NW*32) {
            int r = u >> 2, ch = u & 3;
            async_copy16(lds_addr(&As[buf][r*LDK + ch*8]),
                         A + (long long)(m0 + r)*lda + k0 + ch*8);
        }
    };
    auto stageB = [&](int buf, int k0) {
        if constexpr (!BKN) {
            for (int u = tid; u < BN*4; u += NW*32) {
                int r = u >> 2, ch = u & 3;
                async_copy16(lds_addr(&Bs[buf][r*LDK + ch*8]),
                             B + (long long)(n0 + r)*ldb + k0 + ch*8);
            }
        } else {
            constexpr int NCH = BN/8;
            for (int u = tid; u < 32*NCH; u += NW*32) {
                int kk = u / NCH, nc = u % NCH;
                alignas(16) f16 tmp[8];
                *(u32x4*)tmp =
                    *(const u32x4*)(B + (long long)(k0 + kk)*ldb + n0 + nc*8);
                #pragma unroll
                for (int j = 0; j < 8; ++j) Bs[buf][(nc*8 + j)*LDK + kk] = tmp[j];
            }
        }
    };

    v8f acc[4][2] = {};
    stageA(0, 0);
    stageB(0, 0);
    const int nsteps = K >> 5;

    for (int s = 0; s < nsteps; ++s) {
        wait_async0();            // my async copies into buf[cur] are done
        __syncthreads();          // everyone's copies done; prev compute done
        const int cur = s & 1;
        if (s + 1 < nsteps) {     // prefetch next tile while we compute
            stageA(1 - cur, (s + 1) * 32);
            stageB(1 - cur, (s + 1) * 32);
        }

        Frag af[4], bf[2];
        #pragma unroll
        for (int mt = 0; mt < 4; ++mt) {
            const f16* p = &As[cur][(wm*64 + mt*16 + l16)*LDK + hf*8];
            af[mt].q[0] = *(const u32x4*)p;          // K = hf*8 .. +7
            af[mt].q[1] = *(const u32x4*)(p + 16);   // K = 16+hf*8 .. +7
        }
        #pragma unroll
        for (int nt = 0; nt < 2; ++nt) {
            const f16* p = &Bs[cur][(wn*32 + nt*16 + l16)*LDK + hf*8];
            bf[nt].q[0] = *(const u32x4*)p;          // K = hf*8 .. +7
            bf[nt].q[1] = *(const u32x4*)(p + 16);   // K = 16+hf*8 .. +7
        }
        #pragma unroll
        for (int mt = 0; mt < 4; ++mt)
            #pragma unroll
            for (int nt = 0; nt < 2; ++nt)
                acc[mt][nt] = wmma_f16(af[mt], bf[nt], acc[mt][nt]);
    }

    // ---- epilogue -------------------------------------------------------
    #pragma unroll
    for (int mt = 0; mt < 4; ++mt)
      #pragma unroll
      for (int nt = 0; nt < 2; ++nt)
        #pragma unroll
        for (int r = 0; r < 8; ++r) {
            int row = m0 + wm*64 + mt*16 + r + hf*8;
            int col = n0 + wn*32 + nt*16 + l16;
            float v = acc[mt][nt][r];
            if constexpr (EPI == 0) {
                ((f16*)Cv)[z*sC + (long long)row*ldc + col] = (f16)v;
            } else if constexpr (EPI == 1) {
                float o = ((row == col) ? 2.0f : 0.0f) - v;
                ((f16*)Cv)[z*sC + (long long)row*ldc + col] = (f16)o;
            } else if constexpr (EPI == 2) {
                ((float*)Cv)[z*sC + (long long)row*ldc + col] = v + bias[col];
            } else if constexpr (EPI == 3) {
                ((f16*)Cv)[z*sC + (long long)col*ldc + row] = (f16)v;
            } else if constexpr (EPI == 5) {
                int b_ = row >> 13, n_ = row & 8191;
                int sel = col >> 9, h_ = (col >> 6) & 7, d_ = col & 63;
                long long bh = (long long)(b_*HEADS + h_);
                if (sel == 0)      q_out[(bh*NSEQ + n_)*HDIM + d_] = (f16)(v * SCALEF);
                else if (sel == 1) k_out[(bh*NSEQ + n_)*HDIM + d_] = (f16)v;
                else               v_out[(bh*HDIM + d_)*NSEQ + n_] = (f16)v;  // V^T
            }
        }
}

// ---------------------------------------------------------------------------
// fp32 -> f16 convert
// ---------------------------------------------------------------------------
__global__ void f32_to_f16(const float* __restrict__ src, f16* __restrict__ dst, int n) {
    int i = blockIdx.x*256 + threadIdx.x;
    if (i < n) dst[i] = (f16)src[i];
}

// ---------------------------------------------------------------------------
// Landmark pooling: mean of 32 consecutive rows of q/k  -> q_l, k_l  (f16)
// ---------------------------------------------------------------------------
__global__ void pool_landmarks(const f16* __restrict__ qh, const f16* __restrict__ kh,
                               f16* __restrict__ qlh, f16* __restrict__ klh) {
    int idx = blockIdx.x*256 + threadIdx.x;          // 0 .. 2*16*256*64-1
    int t  = idx >> 18;                              // 0=q, 1=k
    int i  = idx & ((1 << 18) - 1);
    int d  = i & 63, l = (i >> 6) & 255, bh = i >> 14;
    const f16* src = (t ? kh : qh) + ((long long)bh*NSEQ + l*32)*HDIM + d;
    float s = 0.f;
    #pragma unroll
    for (int j = 0; j < 32; ++j) s += (float)src[j*HDIM];
    (t ? klh : qlh)[i] = (f16)(s * (1.f/32.f));
}

// ---------------------------------------------------------------------------
// Row softmax over 256 raw f16 scores -> normalized f16 A (kernel_2)
// ---------------------------------------------------------------------------
__global__ void softmax256(const f16* __restrict__ Sc, f16* __restrict__ Ah) {
    long long row = blockIdx.x;
    const f16* src = Sc + row*NL;
    f16* dst = Ah + row*NL;
    __shared__ float red[64];
    int t = threadIdx.x;
    float v0[4];
    #pragma unroll
    for (int j = 0; j < 4; ++j) v0[j] = (float)src[t + 64*j];
    float mx = fmaxf(fmaxf(v0[0], v0[1]), fmaxf(v0[2], v0[3]));
    red[t] = mx; __syncthreads();
    for (int s = 32; s > 0; s >>= 1) { if (t < s) red[t] = fmaxf(red[t], red[t+s]); __syncthreads(); }
    float m = red[0]; __syncthreads();
    float e[4], sm = 0.f;
    #pragma unroll
    for (int j = 0; j < 4; ++j) { e[j] = __expf(v0[j] - m); sm += e[j]; }
    red[t] = sm; __syncthreads();
    for (int s = 32; s > 0; s >>= 1) { if (t < s) red[t] += red[t+s]; __syncthreads(); }
    float inv = 1.f / red[0];
    #pragma unroll
    for (int j = 0; j < 4; ++j) dst[t + 64*j] = (f16)(e[j] * inv);
}

// ---------------------------------------------------------------------------
// n1 = max column sum of A (deterministic, no atomics); ninf == 1 (softmax)
// ---------------------------------------------------------------------------
__global__ void colsum_n1(const f16* __restrict__ Ah, float* __restrict__ n1inv) {
    int bh = blockIdx.x, c = threadIdx.x;
    const f16* base = Ah + (long long)bh*NL*NL;
    float s = 0.f;
    for (int r = 0; r < NL; ++r) s += (float)base[r*NL + c];
    __shared__ float red[256];
    red[c] = s; __syncthreads();
    for (int st = 128; st > 0; st >>= 1) { if (c < st) red[c] = fmaxf(red[c], red[c+st]); __syncthreads(); }
    if (c == 0) n1inv[bh] = 1.0f / red[0];
}

// Ak0 = A^T / (n1 * ninf),  ninf == 1
__global__ void init_ak(const f16* __restrict__ Ah, const float* __restrict__ n1inv,
                        f16* __restrict__ Ak) {
    int bh = blockIdx.x >> 8, i = blockIdx.x & 255, j = threadIdx.x;
    long long base = (long long)bh*NL*NL;
    Ak[base + i*NL + j] = (f16)((float)Ah[base + j*NL + i] * n1inv[bh]);
}

// ---------------------------------------------------------------------------
// In-place row softmax over 8192 f16 scores (kernel_3)
// ---------------------------------------------------------------------------
__global__ void softmax8192(f16* __restrict__ S) {
    f16* p = S + (long long)blockIdx.x*NSEQ;
    __shared__ float red[256];
    int t = threadIdx.x;
    float mx = -1e30f;
    for (int i = t; i < NSEQ; i += 256) mx = fmaxf(mx, (float)p[i]);
    red[t] = mx; __syncthreads();
    for (int s = 128; s > 0; s >>= 1) { if (t < s) red[t] = fmaxf(red[t], red[t+s]); __syncthreads(); }
    float m = red[0]; __syncthreads();
    float sm = 0.f;
    for (int i = t; i < NSEQ; i += 256) sm += __expf((float)p[i] - m);
    red[t] = sm; __syncthreads();
    for (int s = 128; s > 0; s >>= 1) { if (t < s) red[t] += red[t+s]; __syncthreads(); }
    float inv = 1.f / red[0];
    for (int i = t; i < NSEQ; i += 256) p[i] = (f16)(__expf((float)p[i] - m) * inv);
}

// ---------------------------------------------------------------------------
// Fused: S = q(64xhd) @ k_l^T  -> row softmax (in 64KB LDS, f32)
//        attn = P(64x256) @ R(256x64), R given transposed (Rt[64][256]).
// Output written to attnh[b][n][h*64+d] (merged-head layout, f16).
// ---------------------------------------------------------------------------
__global__ void __launch_bounds__(256)
fused_attn(const f16* __restrict__ qh, const f16* __restrict__ klh,
           const f16* __restrict__ Rt, f16* __restrict__ attnh)
{
    __shared__ alignas(16) float S[64*NL];          // 64 KB
    const int bh = blockIdx.y;
    const int n0 = blockIdx.x * 64;
    const int b_ = bh >> 3, h_ = bh & 7;
    const f16* qp = qh  + (long long)bh*NSEQ*HDIM;
    const f16* kp = klh + (long long)bh*NL*HDIM;
    const f16* rp = Rt  + (long long)bh*HDIM*NL;
    const int tid = threadIdx.x, wave = tid >> 5, lane = tid & 31;
    const int hf = lane >> 4, l16 = lane & 15;

    // GEMM1: all 8 waves share rows 0..63; wave owns cols wave*32..+31.
    v8f acc1[4][2] = {};
    #pragma unroll
    for (int k0 = 0; k0 < HDIM; k0 += 32) {
        Frag af[4], bf[2];
        #pragma unroll
        for (int mt = 0; mt < 4; ++mt) {
            const f16* p = qp + (long long)(n0 + mt*16 + l16)*HDIM + k0 + hf*8;
            af[mt].q[0] = *(const u32x4*)p;
            af[mt].q[1] = *(const u32x4*)(p + 16);
        }
        #pragma unroll
        for (int nt = 0; nt < 2; ++nt) {
            const f16* p = kp + (long long)(wave*32 + nt*16 + l16)*HDIM + k0 + hf*8;
            bf[nt].q[0] = *(const u32x4*)p;
            bf[nt].q[1] = *(const u32x4*)(p + 16);
        }
        #pragma unroll
        for (int mt = 0; mt < 4; ++mt)
            #pragma unroll
            for (int nt = 0; nt < 2; ++nt)
                acc1[mt][nt] = wmma_f16(af[mt], bf[nt], acc1[mt][nt]);
    }
    #pragma unroll
    for (int mt = 0; mt < 4; ++mt)
      #pragma unroll
      for (int nt = 0; nt < 2; ++nt)
        #pragma unroll
        for (int r = 0; r < 8; ++r)
            S[(mt*16 + r + hf*8)*NL + wave*32 + nt*16 + l16] = acc1[mt][nt][r];
    __syncthreads();

    if (tid < 64) {                                   // per-row softmax, in place
        float* row = &S[tid*NL];
        float mx = -1e30f;
        for (int c = 0; c < NL; ++c) mx = fmaxf(mx, row[c]);
        float sm = 0.f;
        for (int c = 0; c < NL; ++c) { float e = __expf(row[c] - mx); row[c] = e; sm += e; }
        float inv = 1.f / sm;
        for (int c = 0; c < NL; ++c) row[c] *= inv;
    }
    __syncthreads();

    // GEMM2: wave tile 16x32 -> 4x2 waves cover 64x64 output.
    const int mrow0 = (wave >> 1) * 16, ncol0 = (wave & 1) * 32;
    v8f acc2[2] = {};
    for (int k0 = 0; k0 < NL; k0 += 32) {
        Frag af;
        const float* srow = &S[(mrow0 + l16)*NL + k0];
        #pragma unroll
        for (int j = 0; j < 8; ++j) {
            af.h[j]     = (f16)srow[hf*8 + j];
            af.h[8 + j] = (f16)srow[16 + hf*8 + j];
        }
        Frag bf[2];
        #pragma unroll
        for (int nt = 0; nt < 2; ++nt) {
            const f16* p = rp + (long long)(ncol0 + nt*16 + l16)*NL + k0 + hf*8;
            bf[nt].q[0] = *(const u32x4*)p;
            bf[nt].q[1] = *(const u32x4*)(p + 16);
        }
        acc2[0] = wmma_f16(af, bf[0], acc2[0]);
        acc2[1] = wmma_f16(af, bf[1], acc2[1]);
    }
    #pragma unroll
    for (int nt = 0; nt < 2; ++nt)
      #pragma unroll
      for (int r = 0; r < 8; ++r) {
          int m = mrow0 + r + hf*8;
          int d = ncol0 + nt*16 + l16;
          attnh[((long long)b_*NSEQ + n0 + m)*DIMC + h_*HDIM + d] = (f16)acc2[nt][r];
      }
}

// ---------------------------------------------------------------------------
// Host-side orchestration
// ---------------------------------------------------------------------------
static inline size_t align256(size_t x) { return (x + 255) & ~size_t(255); }

extern "C" void kernel_launch(void* const* d_in, const int* in_sizes, int n_in,
                              void* d_out, int out_size, void* d_ws, size_t ws_size,
                              hipStream_t stream) {
    (void)in_sizes; (void)n_in; (void)out_size; (void)ws_size;
    const float* x    = (const float*)d_in[0];   // [2,8192,512]
    const float* wqkv = (const float*)d_in[1];   // [1536,512]
    const float* wout = (const float*)d_in[2];   // [512,512]
    const float* bout = (const float*)d_in[3];   // [512]
    float* out = (float*)d_out;                  // [2,8192,512]

    // ---- workspace carve (~166 MB needed) ----
    char* p = (char*)d_ws;
    auto take = [&](size_t bytes) -> char* { char* r = p; p += align256(bytes); return r; };
    f16*   xh    = (f16*)  take((size_t)16384*512*2);
    f16*   wqkvh = (f16*)  take((size_t)1536*512*2);
    f16*   wouth = (f16*)  take((size_t)512*512*2);
    f16*   qh    = (f16*)  take((size_t)BHN*NSEQ*HDIM*2);
    f16*   kh    = (f16*)  take((size_t)BHN*NSEQ*HDIM*2);
    f16*   vt    = (f16*)  take((size_t)BHN*NSEQ*HDIM*2);  // V^T: [bh][hd][n]
    f16*   qlh   = (f16*)  take((size_t)BHN*NL*HDIM*2);
    f16*   klh   = (f16*)  take((size_t)BHN*NL*HDIM*2);
    f16*   Sc    = (f16*)  take((size_t)BHN*NL*NL*2);     // raw kernel_2 scores
    f16*   Ah    = (f16*)  take((size_t)BHN*NL*NL*2);     // softmaxed A
    f16*   Aka   = (f16*)  take((size_t)BHN*NL*NL*2);
    f16*   Akb   = (f16*)  take((size_t)BHN*NL*NL*2);
    f16*   Yh    = (f16*)  take((size_t)BHN*NL*NL*2);
    f16*   S3h   = (f16*)  take((size_t)BHN*NL*NSEQ*2);   // kernel_3 (in-place softmax)
    f16*   T1h   = (f16*)  take((size_t)BHN*NL*HDIM*2);
    f16*   Rt    = (f16*)  take((size_t)BHN*HDIM*NL*2);   // (pinv @ T1)^T
    f16*   attnh = (f16*)  take((size_t)16384*512*2);
    float* n1inv = (float*)take((size_t)BHN*4);

    // 1) fp32 -> f16 conversions
    f32_to_f16<<<dim3(32768), 256, 0, stream>>>(x,    xh,    16384*512);
    f32_to_f16<<<dim3(3072),  256, 0, stream>>>(wqkv, wqkvh, 1536*512);
    f32_to_f16<<<dim3(1024),  256, 0, stream>>>(wout, wouth, 512*512);

    // 2) QKV projection, scatter to q/k [B,H,N,hd] (q scaled) and V^T [B,H,hd,N]
    gemm_wmma<128,128,false,5><<<dim3(128,12,1), 256, 0, stream>>>(
        xh, wqkvh, nullptr, 512, 512, 512, 0, 0, 0, 0, nullptr, qh, kh, vt);

    // 3) landmark pooling
    pool_landmarks<<<dim3(2048), 256, 0, stream>>>(qh, kh, qlh, klh);

    // 4) kernel_2 = softmax(q_l @ k_l^T); n1; Ak0 = A^T/n1
    gemm_wmma<128,128,false,0><<<dim3(2,2,BHN), 256, 0, stream>>>(
        qlh, klh, Sc, 64, 64, 64, NL, (long long)NL*HDIM, (long long)NL*HDIM,
        (long long)NL*NL, nullptr, nullptr, nullptr, nullptr);
    softmax256<<<dim3(BHN*NL), 64, 0, stream>>>(Sc, Ah);
    colsum_n1 <<<dim3(BHN), 256, 0, stream>>>(Ah, n1inv);
    init_ak   <<<dim3(BHN*NL), 256, 0, stream>>>(Ah, n1inv, Aka);

    // 5) Newton-Schulz pinv: Ak = Ak @ (2I - A@Ak), 6 iterations
    f16* cur = Aka; f16* nxt = Akb;
    const long long sNL2 = (long long)NL*NL;
    for (int it = 0; it < PITERS; ++it) {
        gemm_wmma<128,128,true,1><<<dim3(2,2,BHN), 256, 0, stream>>>(
            Ah, cur, Yh, NL, NL, NL, NL, sNL2, sNL2, sNL2, nullptr, nullptr, nullptr, nullptr);
        gemm_wmma<128,128,true,0><<<dim3(2,2,BHN), 256, 0, stream>>>(
            cur, Yh, nxt, NL, NL, NL, NL, sNL2, sNL2, sNL2, nullptr, nullptr, nullptr, nullptr);
        f16* t = cur; cur = nxt; nxt = t;
    }

    // 6) kernel_3 = softmax(q_l @ k^T) over N=8192
    gemm_wmma<128,128,false,0><<<dim3(2,64,BHN), 256, 0, stream>>>(
        qlh, kh, S3h, 64, 64, 64, NSEQ, (long long)NL*HDIM, (long long)NSEQ*HDIM,
        (long long)NL*NSEQ, nullptr, nullptr, nullptr, nullptr);
    softmax8192<<<dim3(BHN*NL), 256, 0, stream>>>(S3h);

    // 7) T1 = kernel_3 @ v   [m,hd]   (B = V^T, natural [N][K] -> async path)
    gemm_wmma<256,64,false,0><<<dim3(1,1,BHN), 256, 0, stream>>>(
        S3h, vt, T1h, NSEQ, NSEQ, NSEQ, HDIM, (long long)NL*NSEQ,
        (long long)NSEQ*HDIM, (long long)NL*HDIM, nullptr, nullptr, nullptr, nullptr);

    // 8) Rt = (pinv @ T1)^T  stored [hd][m] so fused kernel reads B contiguously
    gemm_wmma<256,64,true,3><<<dim3(1,1,BHN), 256, 0, stream>>>(
        cur, T1h, Rt, NL, NL, HDIM, NL, sNL2, (long long)NL*HDIM,
        (long long)HDIM*NL, nullptr, nullptr, nullptr, nullptr);

    // 9) attn = softmax(q @ k_l^T) @ R  (fused), merged-head layout
    fused_attn<<<dim3(NSEQ/64, BHN), 256, 0, stream>>>(qh, klh, Rt, attnh);

    // 10) out = attn @ w_out^T + b_out   (f32 output)
    gemm_wmma<128,128,false,2><<<dim3(128,4,1), 256, 0, stream>>>(
        attnh, wouth, out, 512, 512, 512, 512, 0, 0, 0, bout, nullptr, nullptr, nullptr);
}
